// GNN_61692910240507
// MI455X (gfx1250) — matrix operand order repaired and checked by reading, not compiled, not run on hardware
//
#include <hip/hip_runtime.h>
#include <hip/hip_bf16.h>

typedef __attribute__((ext_vector_type(16))) _Float16 v16h;
typedef __attribute__((ext_vector_type(8)))  float    v8f;

#define N_NODES 100000
#define N_EDGES 1600000
#define IN_DIM  128
#define OUT_DIM 64
#define NEG_SLOPE 0.2f
#define ALPHA 1.0f

__device__ __forceinline__ float lrelu(float x) { return x > 0.0f ? x : NEG_SLOPE * x; }

// Monotone float->uint key: preserves ordering so atomicMax(uint) == float max.
__device__ __forceinline__ unsigned fkey(float x) {
    unsigned u = __float_as_uint(x);
    return (u & 0x80000000u) ? ~u : (u | 0x80000000u);
}
__device__ __forceinline__ float funkey(unsigned k) {
    return __uint_as_float((k & 0x80000000u) ? (k & 0x7FFFFFFFu) : ~k);
}

// ---------------------------------------------------------------------------
// 0) Pre-pack a weight matrix W[128, C] (row-major f32) into f16 fragments in
//    exact WMMA-B operand order:
//      frag index t = (ct*4 + kk)*32 + lane,  element j (0..15):
//      value = W[K, ct*16 + (lane&15)]  with  K = kk*32 + (j>>3)*16 + (lane>>4)*8 + (j&7)
//    One thread per fragment -> one contiguous 32B store.
__global__ __launch_bounds__(256) void k_pack_w(const float* __restrict__ W, int C,
                                                _Float16* __restrict__ outp) {
    int t = blockIdx.x * blockDim.x + threadIdx.x;
    int total = (C / 16) * 4 * 32;
    if (t >= total) return;
    int lane = t & 31;
    int kk   = (t >> 5) & 3;
    int ct   = t >> 7;
    int half = lane >> 4;
    int col  = ct * 16 + (lane & 15);
    v16h v;
    #pragma unroll
    for (int j = 0; j < 16; ++j) {
        int K = kk * 32 + (j >> 3) * 16 + half * 8 + (j & 7);
        v[j] = (_Float16)W[(size_t)K * C + col];
    }
    ((v16h*)outp)[t] = v;
}

// ---------------------------------------------------------------------------
// 1) degree init (self-loop counts as 1)
__global__ __launch_bounds__(256) void k_node_init(float* __restrict__ degf) {
    int n = blockIdx.x * blockDim.x + threadIdx.x;
    if (n < N_NODES) degf[n] = 1.0f;
}

// 2) in-degree over edges
__global__ __launch_bounds__(256) void k_edge_deg(const int* __restrict__ ei,
                                                  float* __restrict__ degf) {
    int e = blockIdx.x * blockDim.x + threadIdx.x;
    if (e < N_EDGES) atomicAdd(&degf[ei[N_EDGES + e]], 1.0f);
}

// ---------------------------------------------------------------------------
// 3) Fused dual WMMA GEMM: h = x @ W_gcn, g = x @ W_gat.
//    Block = 256 threads = 8 waves; block owns 16 rows; wave w owns col tile w.
//    A tile (16x128) staged once into LDS in WMMA-A operand order as f16;
//    B fragments are single contiguous 32B loads from the packed weights.
__global__ __launch_bounds__(256) void k_gemm_xw2(const float* __restrict__ x,
                                                  const _Float16* __restrict__ Wp1,
                                                  const _Float16* __restrict__ Wp2,
                                                  float* __restrict__ h,
                                                  float* __restrict__ g) {
    __shared__ v16h sA[128];                    // 4 k-steps x 32 lanes, 4 KB
    const int rowTile = blockIdx.x;             // 6250 tiles of 16 rows
    const int tid = threadIdx.x;

    // Stage A: thread t fills half a fragment (8 contiguous f16 = 16B).
    {
        int slot = tid >> 1, sub = tid & 1;     // slot = kk*32 + lane
        int kslot = slot >> 5, lslot = slot & 31;
        int row = rowTile * 16 + (lslot & 15);
        int halfs = lslot >> 4;
        const float* src = x + (size_t)row * IN_DIM + kslot * 32 + sub * 16 + halfs * 8;
        _Float16* d = (_Float16*)sA + tid * 8;
        #pragma unroll
        for (int j = 0; j < 8; ++j) d[j] = (_Float16)src[j];
    }
    __syncthreads();

    const int wave = tid >> 5;                  // col tile 0..7
    const int lane = tid & 31;
    const int half = lane >> 4;
    const int col0 = wave * 16;
    const v16h* B1 = (const v16h*)Wp1 + (size_t)wave * 4 * 32 + lane;
    const v16h* B2 = (const v16h*)Wp2 + (size_t)wave * 4 * 32 + lane;
    v8f acc1 = {}, acc2 = {};
    #pragma unroll
    for (int kk = 0; kk < 4; ++kk) {
        v16h a  = sA[kk * 32 + lane];
        v16h b1 = B1[(size_t)kk * 32];
        v16h b2 = B2[(size_t)kk * 32];
        acc1 = __builtin_amdgcn_wmma_f32_16x16x32_f16(false, a, false, b1, (short)0, acc1,
                                                      false, false);
        acc2 = __builtin_amdgcn_wmma_f32_16x16x32_f16(false, a, false, b2, (short)0, acc2,
                                                      false, false);
    }
    float* oh = h + col0 + (lane & 15);
    float* og = g + col0 + (lane & 15);
    #pragma unroll
    for (int r = 0; r < 8; ++r) {
        size_t ro = (size_t)(rowTile * 16 + r + half * 8) * IN_DIM;
        oh[ro] = acc1[r];
        og[ro] = acc2[r];
    }
}

// ---------------------------------------------------------------------------
// 4) per-node attention logits + dinv + self-loop max seed. One wave per node.
__global__ __launch_bounds__(256) void k_node_att(const float* __restrict__ g,
                                                  const float* __restrict__ att_src,
                                                  const float* __restrict__ att_dst,
                                                  const float* __restrict__ degf,
                                                  float* __restrict__ a_s,
                                                  float* __restrict__ a_d,
                                                  float* __restrict__ dinv,
                                                  unsigned* __restrict__ ekey) {
    int node = (blockIdx.x * blockDim.x + threadIdx.x) >> 5;
    int lane = threadIdx.x & 31;
    if (node >= N_NODES) return;
    float4 gv = *(const float4*)(g + (size_t)node * IN_DIM + lane * 4);
    float4 s4 = *(const float4*)(att_src + lane * 4);
    float4 d4 = *(const float4*)(att_dst + lane * 4);
    float s = gv.x * s4.x + gv.y * s4.y + gv.z * s4.z + gv.w * s4.w;
    float d = gv.x * d4.x + gv.y * d4.y + gv.z * d4.z + gv.w * d4.w;
    #pragma unroll
    for (int off = 16; off > 0; off >>= 1) {
        s += __shfl_down(s, off);
        d += __shfl_down(d, off);
    }
    if (lane == 0) {
        a_s[node] = s;
        a_d[node] = d;
        dinv[node] = rsqrtf(degf[node]);
        ekey[node] = fkey(lrelu(s + d));   // self-loop seeds the max
    }
}

// 5) segment max over incoming edges (native u32 atomic max on monotone key)
__global__ __launch_bounds__(256) void k_edge_max(const int* __restrict__ ei,
                                                  const float* __restrict__ a_s,
                                                  const float* __restrict__ a_d,
                                                  unsigned* __restrict__ ekey) {
    int e = blockIdx.x * blockDim.x + threadIdx.x;
    if (e >= N_EDGES) return;
    int src = ei[e], dst = ei[N_EDGES + e];
    atomicMax(&ekey[dst], fkey(lrelu(a_s[src] + a_d[dst])));
}

// 6) per-node: decode max, seed denominators and accumulators with self-loop terms.
//    One wave per node, lane owns 4 features.
__global__ __launch_bounds__(256) void k_node_prep(const float* __restrict__ h,
                                                   const float* __restrict__ g,
                                                   const float* __restrict__ a_s,
                                                   const float* __restrict__ a_d,
                                                   const unsigned* __restrict__ ekey,
                                                   const float* __restrict__ dinv,
                                                   float* __restrict__ emax,
                                                   float* __restrict__ den,
                                                   float* __restrict__ agg_gcn,
                                                   float* __restrict__ agg_gat) {
    int node = (blockIdx.x * blockDim.x + threadIdx.x) >> 5;
    int lane = threadIdx.x & 31;
    if (node >= N_NODES) return;
    float em  = funkey(ekey[node]);
    float exs = expf(lrelu(a_s[node] + a_d[node]) - em);
    float dv  = dinv[node];
    float snorm = dv * dv;
    if (lane == 0) { emax[node] = em; den[node] = exs; }
    float4 hv = *(const float4*)(h + (size_t)node * IN_DIM + lane * 4);
    float4 gv = *(const float4*)(g + (size_t)node * IN_DIM + lane * 4);
    float4 ag = make_float4(hv.x * snorm, hv.y * snorm, hv.z * snorm, hv.w * snorm);
    float4 at = make_float4(gv.x * exs, gv.y * exs, gv.z * exs, gv.w * exs);
    *(float4*)(agg_gcn + (size_t)node * IN_DIM + lane * 4) = ag;
    *(float4*)(agg_gat + (size_t)node * IN_DIM + lane * 4) = at;
}

// 7) heavy phase: wave32 per edge; lane owns 4 features; 8 f32 atomics per lane.
//    Softmax normalization deferred (numerator + denominator accumulated together).
__global__ __launch_bounds__(256) void k_edge_agg(const int* __restrict__ ei,
                                                  const float* __restrict__ h,
                                                  const float* __restrict__ g,
                                                  const float* __restrict__ a_s,
                                                  const float* __restrict__ a_d,
                                                  const float* __restrict__ emax,
                                                  const float* __restrict__ dinv,
                                                  float* __restrict__ den,
                                                  float* __restrict__ agg_gcn,
                                                  float* __restrict__ agg_gat) {
    int edge = (blockIdx.x * blockDim.x + threadIdx.x) >> 5;
    int lane = threadIdx.x & 31;
    if (edge >= N_EDGES) return;
    int src = ei[edge], dst = ei[N_EDGES + edge];
    float ex   = expf(lrelu(a_s[src] + a_d[dst]) - emax[dst]);    // wave-uniform
    float coef = dinv[src] * dinv[dst];                           // wave-uniform
    if (lane == 0) atomicAdd(&den[dst], ex);
    float4 hv = *(const float4*)(h + (size_t)src * IN_DIM + lane * 4);
    float4 gv = *(const float4*)(g + (size_t)src * IN_DIM + lane * 4);
    float* pg = agg_gcn + (size_t)dst * IN_DIM + lane * 4;
    float* pa = agg_gat + (size_t)dst * IN_DIM + lane * 4;
    atomicAdd(pg + 0, hv.x * coef); atomicAdd(pg + 1, hv.y * coef);
    atomicAdd(pg + 2, hv.z * coef); atomicAdd(pg + 3, hv.w * coef);
    atomicAdd(pa + 0, gv.x * ex);   atomicAdd(pa + 1, gv.y * ex);
    atomicAdd(pa + 2, gv.z * ex);   atomicAdd(pa + 3, gv.w * ex);
}

// ---------------------------------------------------------------------------
// 8) fused combine: out = tanh((agg_gcn+b1) @ w1 + ALPHA*(agg_gat/den + b2) @ w2)
//    Both GEMMs share one f32 accumulator: 8 WMMAs per 16x16 tile. Packed-f16 B.
__global__ __launch_bounds__(128) void k_final(const float* __restrict__ agg_gcn,
                                               const float* __restrict__ agg_gat,
                                               const float* __restrict__ den,
                                               const float* __restrict__ b_gcn,
                                               const float* __restrict__ b_gat,
                                               const _Float16* __restrict__ w1p,
                                               const _Float16* __restrict__ w2p,
                                               float* __restrict__ out) {
    const int rowTile = blockIdx.x;            // 6250 tiles of 16 rows
    const int wave = threadIdx.x >> 5;         // 0..3 -> col tile (64 cols)
    const int lane = threadIdx.x & 31;
    const int half = lane >> 4;
    const int row  = rowTile * 16 + (lane & 15);
    const int col0 = wave * 16;
    const float rden = 1.0f / den[row];
    const v16h* B1 = (const v16h*)w1p + (size_t)wave * 4 * 32 + lane;
    const v16h* B2 = (const v16h*)w2p + (size_t)wave * 4 * 32 + lane;
    v8f acc = {};
    #pragma unroll
    for (int kk = 0; kk < 4; ++kk) {
        const int kbase = kk * 32;
        v16h a1, a2;
        const float* r1 = agg_gcn + (size_t)row * IN_DIM + kbase;
        const float* r2 = agg_gat + (size_t)row * IN_DIM + kbase;
        const float* bb1 = b_gcn + kbase;
        const float* bb2 = b_gat + kbase;
        #pragma unroll
        for (int e = 0; e < 8; ++e) {
            int k0 = half * 8 + e;
            int k1 = 16 + half * 8 + e;
            a1[e]     = (_Float16)(r1[k0] + bb1[k0]);
            a1[8 + e] = (_Float16)(r1[k1] + bb1[k1]);
            a2[e]     = (_Float16)(ALPHA * (r2[k0] * rden + bb2[k0]));
            a2[8 + e] = (_Float16)(ALPHA * (r2[k1] * rden + bb2[k1]));
        }
        v16h b1 = B1[(size_t)kk * 32];
        v16h b2 = B2[(size_t)kk * 32];
        acc = __builtin_amdgcn_wmma_f32_16x16x32_f16(false, a1, false, b1, (short)0, acc,
                                                     false, false);
        acc = __builtin_amdgcn_wmma_f32_16x16x32_f16(false, a2, false, b2, (short)0, acc,
                                                     false, false);
    }
    float* o = out + col0 + (lane & 15);
    #pragma unroll
    for (int r = 0; r < 8; ++r)
        o[(size_t)(rowTile * 16 + r + half * 8) * OUT_DIM] = tanhf(acc[r]);
}

// ---------------------------------------------------------------------------
extern "C" void kernel_launch(void* const* d_in, const int* in_sizes, int n_in,
                              void* d_out, int out_size, void* d_ws, size_t ws_size,
                              hipStream_t stream) {
    const float* x       = (const float*)d_in[0];
    const int*   ei      = (const int*)d_in[1];
    const float* W_gcn   = (const float*)d_in[2];
    const float* b_gcn   = (const float*)d_in[3];
    const float* W_gat   = (const float*)d_in[4];
    const float* att_src = (const float*)d_in[5];
    const float* att_dst = (const float*)d_in[6];
    const float* b_gat   = (const float*)d_in[7];
    const float* w_gcn   = (const float*)d_in[8];
    const float* w_gat   = (const float*)d_in[9];
    float* out = (float*)d_out;

    char* ws = (char*)d_ws;
    size_t off = 0;
    auto carve = [&](size_t bytes) -> void* {
        void* p = ws + off;
        off = (off + bytes + 255) & ~(size_t)255;
        return p;
    };
    float*     degf    = (float*)carve(sizeof(float) * N_NODES);
    float*     dinv    = (float*)carve(sizeof(float) * N_NODES);
    float*     a_s     = (float*)carve(sizeof(float) * N_NODES);
    float*     a_d     = (float*)carve(sizeof(float) * N_NODES);
    unsigned*  ekey    = (unsigned*)carve(sizeof(unsigned) * N_NODES);
    float*     emax    = (float*)carve(sizeof(float) * N_NODES);
    float*     den     = (float*)carve(sizeof(float) * N_NODES);
    _Float16*  Wp_gcn  = (_Float16*)carve(sizeof(_Float16) * IN_DIM * IN_DIM);
    _Float16*  Wp_gat  = (_Float16*)carve(sizeof(_Float16) * IN_DIM * IN_DIM);
    _Float16*  wp_gcn  = (_Float16*)carve(sizeof(_Float16) * IN_DIM * OUT_DIM);
    _Float16*  wp_gat  = (_Float16*)carve(sizeof(_Float16) * IN_DIM * OUT_DIM);
    float*     h       = (float*)carve(sizeof(float) * (size_t)N_NODES * IN_DIM);
    float*     g       = (float*)carve(sizeof(float) * (size_t)N_NODES * IN_DIM);
    float*     agg_gcn = (float*)carve(sizeof(float) * (size_t)N_NODES * IN_DIM);
    float*     agg_gat = (float*)carve(sizeof(float) * (size_t)N_NODES * IN_DIM);

    const int ROW_TILES = N_NODES / 16;  // 6250, exact

    // weight pre-pack (tiny)
    k_pack_w<<<4, 256, 0, stream>>>(W_gcn, IN_DIM, Wp_gcn);
    k_pack_w<<<4, 256, 0, stream>>>(W_gat, IN_DIM, Wp_gat);
    k_pack_w<<<2, 256, 0, stream>>>(w_gcn, OUT_DIM, wp_gcn);
    k_pack_w<<<2, 256, 0, stream>>>(w_gat, OUT_DIM, wp_gat);

    k_node_init<<<(N_NODES + 255) / 256, 256, 0, stream>>>(degf);
    k_edge_deg<<<(N_EDGES + 255) / 256, 256, 0, stream>>>(ei, degf);
    k_gemm_xw2<<<ROW_TILES, 256, 0, stream>>>(x, Wp_gcn, Wp_gat, h, g);
    k_node_att<<<((size_t)N_NODES * 32 + 255) / 256, 256, 0, stream>>>(
        g, att_src, att_dst, degf, a_s, a_d, dinv, ekey);
    k_edge_max<<<(N_EDGES + 255) / 256, 256, 0, stream>>>(ei, a_s, a_d, ekey);
    k_node_prep<<<((size_t)N_NODES * 32 + 255) / 256, 256, 0, stream>>>(
        h, g, a_s, a_d, ekey, dinv, emax, den, agg_gcn, agg_gat);
    k_edge_agg<<<((size_t)N_EDGES * 32 + 255) / 256, 256, 0, stream>>>(
        ei, h, g, a_s, a_d, emax, dinv, den, agg_gcn, agg_gat);
    k_final<<<ROW_TILES, 128, 0, stream>>>(agg_gcn, agg_gat, den, b_gcn, b_gat,
                                           wp_gcn, wp_gat, out);
}
